// ConvMamba_58884001628553
// MI455X (gfx1250) — matrix operand order repaired and checked by reading, not compiled, not run on hardware
//
#include <hip/hip_runtime.h>
#include <hip/hip_bf16.h>
#include <math.h>

// Problem constants (match reference)
#define BB      64
#define SS      256
#define DD      128
#define DI_     256
#define DSTATE  16
#define DTR     8
#define NTOK    (BB * SS)        // 16384
#define XP_N    (DTR + 2*DSTATE) // 40
#define XZ_N    (2 * DI_)        // 512
#define W2COLS  (9 * DD)         // 1152 (transposed conv2d weight row length)

typedef __attribute__((ext_vector_type(16))) __bf16 v16bf;
typedef __attribute__((ext_vector_type(8)))  float  v8f;
typedef __attribute__((ext_vector_type(4)))  unsigned int v4u;
typedef __attribute__((ext_vector_type(8)))  int    v8i;
typedef __attribute__((ext_vector_type(4)))  int    v4i;

#if defined(__has_builtin)
#if __has_builtin(__builtin_amdgcn_tensor_load_to_lds) && __has_builtin(__builtin_amdgcn_s_wait_tensorcnt)
#define HAVE_TDM 1
#endif
#endif

__device__ __forceinline__ float silu_f(float x) { return x / (1.f + __expf(-x)); }

// Pack 8 consecutive f32 (two float4) into bf16 fragment slots [base..base+7].
__device__ __forceinline__ void pack8(v16bf& f, int base, float4 u, float4 v) {
  f[base+0] = (__bf16)u.x; f[base+1] = (__bf16)u.y;
  f[base+2] = (__bf16)u.z; f[base+3] = (__bf16)u.w;
  f[base+4] = (__bf16)v.x; f[base+5] = (__bf16)v.y;
  f[base+6] = (__bf16)v.z; f[base+7] = (__bf16)v.w;
}
__device__ __forceinline__ void pack8s(v16bf& f, int base, float4 u, float4 v, float s) {
  f[base+0] = (__bf16)(u.x*s); f[base+1] = (__bf16)(u.y*s);
  f[base+2] = (__bf16)(u.z*s); f[base+3] = (__bf16)(u.w*s);
  f[base+4] = (__bf16)(v.x*s); f[base+5] = (__bf16)(v.y*s);
  f[base+6] = (__bf16)(v.z*s); f[base+7] = (__bf16)(v.w*s);
}

// Load one contiguous 16-float run as a bf16 B/A fragment half-pair.
__device__ __forceinline__ v16bf frag16(const float* p) {
  const float4* q = (const float4*)p;
  float4 b0 = q[0], b1 = q[1], b2 = q[2], b3 = q[3];
  v16bf f; pack8(f, 0, b0, b1); pack8(f, 8, b2, b3);
  return f;
}

// ---------------------------------------------------------------------------
// TDM: stage a 2D f32 panel (rows x rowElems, row stride strideElems) from
// global into LDS at byte offset ldsOff. Rows >= rowsAvail are zero-filled by
// the TDM OOB rule (tensor_dim1). D# per CDNA5 ISA.
// ---------------------------------------------------------------------------
#ifdef HAVE_TDM
__device__ __forceinline__ void tdm_load_panel_f32(const float* gsrc, unsigned ldsOff,
                                                   int rowElems, int rows,
                                                   int rowsAvail, int strideElems) {
  unsigned long long ga = (unsigned long long)(size_t)gsrc;
  v4u g0;
  g0[0] = 1u;                                        // count=1 (valid), user mode
  g0[1] = ldsOff;                                    // lds_addr (bytes)
  g0[2] = (unsigned)(ga & 0xffffffffull);            // global_addr[31:0]
  g0[3] = (unsigned)((ga >> 32) & 0x1ffffffull)      // global_addr[56:32]
        | (2u << 30);                                // type = 2 ("image")
  v8i g1;
  g1[0] = (int)(2u << 16);                           // data_size = 2 -> 4 bytes
  g1[1] = (int)((unsigned)(rowElems & 0xffff) << 16);           // tensor_dim0[15:0]
  g1[2] = (int)(((unsigned)(rowsAvail & 0xffff) << 16)          // tensor_dim1[15:0]
              | (((unsigned)rowElems >> 16) & 0xffffu));        // tensor_dim0[31:16]
  g1[3] = (int)(((unsigned)(rowElems & 0xffff) << 16)           // tile_dim0
              | (((unsigned)rowsAvail >> 16) & 0xffffu));       // tensor_dim1[31:16]
  g1[4] = rows & 0xffff;                             // tile_dim1 (tile_dim2 = 0)
  g1[5] = strideElems;                               // tensor_dim0_stride[31:0]
  g1[6] = 0;                                         // stride hi / dim1_stride lo
  g1[7] = 0;
  v4i gz = {0, 0, 0, 0};
#if __clang_major__ >= 23
  v8i gz8 = {0, 0, 0, 0, 0, 0, 0, 0};
  __builtin_amdgcn_tensor_load_to_lds(g0, g1, gz, gz, gz8, 0);
#else
  __builtin_amdgcn_tensor_load_to_lds(g0, g1, gz, gz, 0);
#endif
}
#endif

// ---------------------------------------------------------------------------
// GEMM: C[M,N] = A[M,K] * W[N,K]^T  (bf16 WMMA, f32 accumulate)
// Compile-time K and N. block = 128 threads = 4 waves covering a 64-row by
// 32-col C panel: each wave owns one 16-row strip and BOTH 16-col tiles,
// reusing its A fragment for 2 WMMA per k-step (independent accumulators).
// B panel (32 x K) staged once per block via TDM (zero-filled if N ragged).
// Requirements: M % 64 == 0, K % 32 == 0, K <= 256.
// ---------------------------------------------------------------------------
template <int K, int N>
__global__ void cm_gemm_wmma_t(const float* __restrict__ A, const float* __restrict__ W,
                               float* __restrict__ C, int M) {
  __shared__ float sB[32 * K];
  const int n0 = blockIdx.x * 32;
#ifdef HAVE_TDM
  if (threadIdx.x < 32) {                       // one wave issues the DMA
    tdm_load_panel_f32(W + (size_t)n0 * K, (unsigned)(size_t)(&sB[0]),
                       K, 32, N - n0, K);
    __builtin_amdgcn_s_wait_tensorcnt(0);
  }
#else
  constexpr int kq = K >> 2;
  for (int e = threadIdx.x; e < 32 * kq; e += blockDim.x) {
    const int r = e / kq, cc = e - r * kq;
    float4 v = {0.f, 0.f, 0.f, 0.f};
    if (n0 + r < N) v = ((const float4*)(W + (size_t)(n0 + r) * K))[cc];
    ((float4*)sB)[e] = v;
  }
#endif
  __syncthreads();

  const int wave = threadIdx.x >> 5;
  const int lane = threadIdx.x & 31;
  const int m    = (blockIdx.y * 4 + wave) * 16 + (lane & 15);
  const int half = lane >> 4;
  const float* arow  = A + (size_t)m * K;
  const float* brow0 = sB + (size_t)(lane & 15) * K;        // n-tile 0
  const float* brow1 = sB + (size_t)((lane & 15) + 16) * K; // n-tile 1

  v8f c0 = {}, c1 = {};
#pragma unroll
  for (int k0 = 0; k0 < K; k0 += 32) {
    if (k0 + 64 <= K) __builtin_prefetch(arow + k0 + 64, 0, 1); // global_prefetch_b8
    // A fragment: one contiguous 16-float run per lane (k = 16*half + s) -> 4x b128 loads
    const float4* pa  = (const float4*)(arow + k0 + 16 * half);
    const float4* pa2 = (const float4*)(arow + k0 + 16 * half + 8);
    float4 a0 = pa[0], a1 = pa[1], a2 = pa2[0], a3 = pa2[1];
    v16bf af; pack8(af, 0, a0, a1); pack8(af, 8, a2, a3);
    // Two B fragments from the LDS panel (16 contiguous floats each)
    v16bf bf0 = frag16(brow0 + k0 + 16 * half);
    v16bf bf1 = frag16(brow1 + k0 + 16 * half);
    c0 = __builtin_amdgcn_wmma_f32_16x16x32_bf16(false, af, false, bf0, (short)0, c0,
                                                 false, false);
    c1 = __builtin_amdgcn_wmma_f32_16x16x32_bf16(false, af, false, bf1, (short)0, c1,
                                                 false, false);
  }
  // C/D layout: lane -> N = lane%16, M = v + 8*(lane/16)
  const int mh = (blockIdx.y * 4 + wave) * 16 + 8 * half;
  const int nA = n0 + (lane & 15);
  const int nBcol = nA + 16;
  if (nA < N) {
#pragma unroll
    for (int v = 0; v < 8; ++v) C[(size_t)(mh + v) * N + nA] = c0[v];
  }
  if (nBcol < N) {
#pragma unroll
    for (int v = 0; v < 8; ++v) C[(size_t)(mh + v) * N + nBcol] = c1[v];
  }
}

// ---------------------------------------------------------------------------
// Depthwise causal conv1d over sequence + SiLU. xz row = [xc(256) | z(256)].
// ---------------------------------------------------------------------------
__global__ void cm_conv1d_silu(const float* __restrict__ xz, const float* __restrict__ cw,
                               const float* __restrict__ cb, float* __restrict__ xc) {
  const int idx = blockIdx.x * blockDim.x + threadIdx.x;   // t*256 + d
  if (idx >= NTOK * DI_) return;
  const int d = idx & (DI_ - 1);
  const int t = idx >> 8;
  const int l = t & (SS - 1);
  float acc = cb[d];
#pragma unroll
  for (int k = 0; k < 4; ++k) {
    const int ls = l - 3 + k;
    if (ls >= 0) acc += xz[(size_t)(t - (3 - k)) * XZ_N + d] * cw[d * 4 + k];
  }
  xc[idx] = silu_f(acc);
}

// dt = softplus(x_dbl[:, :8] @ Wdt^T + bdt)   (K=8: VALU beats WMMA)
__global__ void cm_dt(const float* __restrict__ xdbl, const float* __restrict__ Wdt,
                      const float* __restrict__ bdt, float* __restrict__ dtb) {
  const int idx = blockIdx.x * blockDim.x + threadIdx.x;
  if (idx >= NTOK * DI_) return;
  const int d = idx & (DI_ - 1);
  const int t = idx >> 8;
  const float4* xr = (const float4*)(xdbl + (size_t)t * XP_N);
  const float4* wr = (const float4*)(Wdt + (size_t)d * DTR);
  float4 x0 = xr[0], x1 = xr[1], w0 = wr[0], w1 = wr[1];
  float acc = bdt[d] + x0.x*w0.x + x0.y*w0.y + x0.z*w0.z + x0.w*w0.w
                     + x1.x*w1.x + x1.y*w1.y + x1.z*w1.z + x1.w*w1.w;
  dtb[idx] = (acc > 20.f) ? acc : log1pf(__expf(acc));
}

// ---------------------------------------------------------------------------
// Selective scan: one block per batch image (b), 256 threads = DI channels.
// B/C staged in LDS in 64-step chunks (shared by all 256 channels).
// Fused: y = (sum_n h*C) + xc*D, then y *= silu(z).
// ---------------------------------------------------------------------------
__global__ void cm_scan(const float* __restrict__ xz, const float* __restrict__ xc,
                        const float* __restrict__ xdbl, const float* __restrict__ dtb,
                        const float* __restrict__ A_log, const float* __restrict__ Dskip,
                        float* __restrict__ y) {
  __shared__ float sBC[64][32];
  const int b = blockIdx.x;
  const int d = threadIdx.x;
  float Ad[DSTATE];
#pragma unroll
  for (int n = 0; n < DSTATE; ++n) Ad[n] = -__expf(A_log[d * DSTATE + n]);
  const float Dp = Dskip[d];
  float h[DSTATE];
#pragma unroll
  for (int n = 0; n < DSTATE; ++n) h[n] = 0.f;

  for (int lc = 0; lc < SS; lc += 64) {
    __syncthreads();
    for (int e = threadIdx.x; e < 64 * 8; e += 256) {   // 512 float4 stages
      const int l = e >> 3, j = e & 7;
      ((float4*)&sBC[l][0])[j] =
          *(const float4*)(xdbl + (size_t)(b * SS + lc + l) * XP_N + DTR + 4 * j);
    }
    __syncthreads();
    for (int l = 0; l < 64; ++l) {
      const size_t t  = (size_t)b * SS + lc + l;
      const float dtv = dtb[t * DI_ + d];
      const float xv  = xc[t * DI_ + d];
      float acc = 0.f;
#pragma unroll
      for (int n = 0; n < DSTATE; ++n) {
        h[n] = __expf(dtv * Ad[n]) * h[n] + dtv * sBC[l][n] * xv;
        acc += h[n] * sBC[l][16 + n];
      }
      acc += xv * Dp;
      const float zv = xz[t * XZ_N + DI_ + d];
      y[t * DI_ + d] = acc * silu_f(zv);
    }
  }
}

// x1 = x_in + silu(layernorm(h))
__global__ void cm_ln_silu_res(const float* __restrict__ h, const float* __restrict__ w,
                               const float* __restrict__ bia, const float* __restrict__ xin,
                               float* __restrict__ x1) {
  const int t = blockIdx.x * blockDim.x + threadIdx.x;
  if (t >= NTOK) return;
  const float* row = h + (size_t)t * DD;
  float mu = 0.f;
  for (int i = 0; i < DD; ++i) mu += row[i];
  mu *= (1.f / DD);
  float var = 0.f;
  for (int i = 0; i < DD; ++i) { const float dlt = row[i] - mu; var += dlt * dlt; }
  var *= (1.f / DD);
  const float inv = rsqrtf(var + 1e-5f);
  for (int i = 0; i < DD; ++i) {
    const float v = (row[i] - mu) * inv * w[i] + bia[i];
    x1[(size_t)t * DD + i] = xin[(size_t)t * DD + i] + silu_f(v);
  }
}

// Pre-transpose conv2d weights HWIO (tap,cin,co) -> Wt[co][tap*128+cin]
__global__ void cm_wt_transpose(const float* __restrict__ Wc, float* __restrict__ Wt) {
  const int idx = blockIdx.x * blockDim.x + threadIdx.x;
  if (idx >= 9 * DD * DD) return;
  const int co = idx & (DD - 1);
  const int ci = (idx >> 7) & (DD - 1);
  const int tap = idx >> 14;
  Wt[(size_t)co * W2COLS + tap * DD + ci] = Wc[idx];
}

// ---------------------------------------------------------------------------
// 3x3 SAME conv over 16x16 image as implicit GEMM, fused bias + residual.
// Each wave: 16 tokens (one image row; same b,y -> tap row-validity is
// wave-uniform) x 32 cout (two n-tiles, A fragment reused for 2 WMMA).
// x-edge lanes use clamped address + zero scale (branchless b128 loads).
// ---------------------------------------------------------------------------
__global__ void cm_conv2d_wmma(const float* __restrict__ X1, const float* __restrict__ Wt,
                               const float* __restrict__ bias, const float* __restrict__ Xres,
                               float* __restrict__ Xout) {
  const int wave  = threadIdx.x >> 5;
  const int lane  = threadIdx.x & 31;
  const int mt    = blockIdx.y * 4 + wave;   // 0..1023 token tiles
  const int nt    = blockIdx.x;              // 0..3 cout panels (32 wide)
  const int t0    = mt * 16;
  const int bImg  = t0 >> 8;
  const int yRow  = (t0 >> 4) & 15;
  const int n0    = nt * 32;
  const int xLane = lane & 15;
  const int half  = lane >> 4;
  const float* wrow0 = Wt + (size_t)(n0 + xLane) * W2COLS;
  const float* wrow1 = Wt + (size_t)(n0 + 16 + xLane) * W2COLS;
  v8f c0 = {}, c1 = {};
  for (int dy = -1; dy <= 1; ++dy) {
    const int yy = yRow + dy;
    if (yy < 0 || yy > 15) continue;         // wave-uniform
    for (int dx = -1; dx <= 1; ++dx) {
      const int   xx   = xLane + dx;
      const float s    = (xx >= 0 && xx < 16) ? 1.f : 0.f;
      const int   xcl  = (xx < 0) ? 0 : ((xx > 15) ? 15 : xx);
      const float* src = X1 + (size_t)(((bImg * 16 + yy) * 16) + xcl) * DD;
      const int   tapO = ((dy + 1) * 3 + (dx + 1)) * DD;
#pragma unroll
      for (int k0 = 0; k0 < DD; k0 += 32) {
        const float4* pa  = (const float4*)(src + k0 + 16 * half);
        const float4* pa2 = (const float4*)(src + k0 + 16 * half + 8);
        float4 a0 = pa[0], a1 = pa[1], a2 = pa2[0], a3 = pa2[1];
        v16bf af; pack8s(af, 0, a0, a1, s); pack8s(af, 8, a2, a3, s);
        v16bf bf0 = frag16(wrow0 + tapO + k0 + 16 * half);
        v16bf bf1 = frag16(wrow1 + tapO + k0 + 16 * half);
        c0 = __builtin_amdgcn_wmma_f32_16x16x32_bf16(false, af, false, bf0, (short)0, c0,
                                                     false, false);
        c1 = __builtin_amdgcn_wmma_f32_16x16x32_bf16(false, af, false, bf1, (short)0, c1,
                                                     false, false);
      }
    }
  }
#pragma unroll
  for (int v = 0; v < 8; ++v) {
    const int t = t0 + 8 * half + v;
    const int nA = n0 + xLane;
    Xout[(size_t)t * DD + nA]      = c0[v] + bias[nA]      + Xres[(size_t)t * DD + nA];
    Xout[(size_t)t * DD + nA + 16] = c1[v] + bias[nA + 16] + Xres[(size_t)t * DD + nA + 16];
  }
}

// centers[b,d] = sum_s x[b,s,d] * head_w[s] + head_b
__global__ void cm_head(const float* __restrict__ x, const float* __restrict__ hw,
                        const float* __restrict__ hb, float* __restrict__ out) {
  const int idx = blockIdx.x * blockDim.x + threadIdx.x;   // b*128 + d
  if (idx >= BB * DD) return;
  const int d = idx & (DD - 1);
  const int b = idx >> 7;
  float acc = 0.f;
  for (int s = 0; s < SS; ++s) acc += x[(size_t)((b << 8) + s) * DD + d] * hw[s];
  out[idx] = acc + hb[0];
}

// ---------------------------------------------------------------------------
extern "C" void kernel_launch(void* const* d_in, const int* in_sizes, int n_in,
                              void* d_out, int out_size, void* d_ws, size_t ws_size,
                              hipStream_t stream) {
  (void)in_sizes; (void)n_in; (void)out_size; (void)ws_size;
  const float* x0        = (const float*)d_in[0];
  const float* in_proj_w = (const float*)d_in[1];
  const float* conv_w    = (const float*)d_in[2];
  const float* conv_b    = (const float*)d_in[3];
  const float* x_proj_w  = (const float*)d_in[4];
  const float* dt_proj_w = (const float*)d_in[5];
  const float* dt_proj_b = (const float*)d_in[6];
  const float* A_log     = (const float*)d_in[7];
  const float* D_skip    = (const float*)d_in[8];
  const float* out_proj_w= (const float*)d_in[9];
  const float* ln_w      = (const float*)d_in[10];
  const float* ln_b      = (const float*)d_in[11];
  const float* conv2d_w  = (const float*)d_in[12];
  const float* conv2d_b  = (const float*)d_in[13];
  const float* head_w    = (const float*)d_in[14];
  const float* head_b    = (const float*)d_in[15];

  float* ws = (float*)d_ws;
  size_t off = 0;
  float* xz   = ws + off; off += (size_t)NTOK * XZ_N;
  float* xc   = ws + off; off += (size_t)NTOK * DI_;
  float* xdbl = ws + off; off += (size_t)NTOK * XP_N;
  float* dtb  = ws + off; off += (size_t)NTOK * DI_;
  float* yb   = ws + off; off += (size_t)NTOK * DI_;
  float* hout = ws + off; off += (size_t)NTOK * DD;
  float* x1   = ws + off; off += (size_t)NTOK * DD;
  float* xpA  = ws + off; off += (size_t)NTOK * DD;
  float* xpB  = ws + off; off += (size_t)NTOK * DD;
  float* Wt   = ws + off; off += (size_t)9 * DD * DD;

  float* outx    = (float*)d_out;                 // final x (64,256,128)
  float* centers = outx + (size_t)NTOK * DD;      // (4,64,128)

  const dim3 gemmBlk(128);
  const int  mtb = NTOK / 64;                     // 256 y-blocks of 4 waves

  // One-time conv2d weight transpose (shared across layers)
  cm_wt_transpose<<<(9 * DD * DD) / 256, 256, 0, stream>>>(conv2d_w, Wt);

  const float* xcur = x0;
  for (int i = 0; i < 4; ++i) {
    const float* Wi   = in_proj_w + (size_t)i * XZ_N * DD;
    const float* cwi  = conv_w    + (size_t)i * DI_ * 4;
    const float* cbi  = conv_b    + (size_t)i * DI_;
    const float* Wxi  = x_proj_w  + (size_t)i * XP_N * DI_;
    const float* Wdti = dt_proj_w + (size_t)i * DI_ * DTR;
    const float* bdti = dt_proj_b + (size_t)i * DI_;
    const float* Ali  = A_log     + (size_t)i * DI_ * DSTATE;
    const float* Dsi  = D_skip    + (size_t)i * DI_;
    const float* Woi  = out_proj_w+ (size_t)i * DD * DI_;
    const float* lnwi = ln_w      + (size_t)i * DD;
    const float* lnbi = ln_b      + (size_t)i * DD;
    float* xnext = (i == 3) ? outx : ((i & 1) ? xpB : xpA);

    // 1) xz = x @ Wi^T          (M=16384, N=512, K=128)
    cm_gemm_wmma_t<DD, XZ_N><<<dim3(XZ_N / 32, mtb), gemmBlk, 0, stream>>>(xcur, Wi, xz, NTOK);
    // 2) depthwise conv1d + silu
    cm_conv1d_silu<<<(NTOK * DI_) / 256, 256, 0, stream>>>(xz, cwi, cbi, xc);
    // 3) x_dbl = xc @ Wx^T      (M=16384, N=40 ragged, K=256)
    cm_gemm_wmma_t<DI_, XP_N><<<dim3((XP_N + 31) / 32, mtb), gemmBlk, 0, stream>>>(xc, Wxi, xdbl, NTOK);
    // 4) dt = softplus(...)
    cm_dt<<<(NTOK * DI_) / 256, 256, 0, stream>>>(xdbl, Wdti, bdti, dtb);
    // 5) selective scan (fused y, D-skip, silu(z) gate)
    cm_scan<<<BB, 256, 0, stream>>>(xz, xc, xdbl, dtb, Ali, Dsi, yb);
    // 6) h = y @ Wo^T           (M=16384, N=128, K=256)
    cm_gemm_wmma_t<DI_, DD><<<dim3(DD / 32, mtb), gemmBlk, 0, stream>>>(yb, Woi, hout, NTOK);
    // 7) x1 = x + silu(LN(h))
    cm_ln_silu_res<<<NTOK / 256, 256, 0, stream>>>(hout, lnwi, lnbi, xcur, x1);
    // 8) x_next = conv2d(x1) + bias + x (implicit GEMM WMMA)
    cm_conv2d_wmma<<<dim3(DD / 32, (NTOK / 16) / 4), gemmBlk, 0, stream>>>(x1, Wt, conv2d_b, xcur, xnext);
    // 9) centers[i] = x_next . head_w + head_b
    cm_head<<<(BB * DD) / 256, 256, 0, stream>>>(xnext, head_w, head_b, centers + (size_t)i * BB * DD);

    xcur = xnext;
  }
}
